// MGE_60919816126534
// MI455X (gfx1250) — compile-verified, matrix operand order checked by reading, not run on hardware
//
#include <hip/hip_runtime.h>
#include <hip/hip_bf16.h>
#include <math.h>

#define EPS_BN 1e-5f
#define KK 40
#define KS 12
#define NB 8

typedef __attribute__((ext_vector_type(16))) _Float16 v16h;
typedef __attribute__((ext_vector_type(8)))  _Float16 v8h;
typedef __attribute__((ext_vector_type(8)))  float    v8f;

union Frag16 { v16h v; v8h h[2]; };

// =====================================================================
// WMMA GEMM: Y(MP x N) = Wh(MPpad32 x Kp) * XT(N x Kp)^T  (f16 in, f32 acc)
// Workgroup = 4 waves, tile 32(M) x 256(N); wave tile 32 x 64.
// Weight block (32 x Kp f16) staged once in LDS (cooperative b128 copies,
// +8-half row pad for bank spread); A fragments then come from ds_load_b128,
// B fragments stream from global as two b128 per 16x32 fragment.
// Fragment layouts per ISA 7.12.2: A lane l = row (l&15), halves j<8 ->
// K=hi+j, j>=8 -> K=16+hi+(j-8), hi=(l>>4)*8; B mirrored; C/D f32: VGPR r,
// lane l -> row r+(l>>4)*8, col l&15.  8 WMMA per K-step per wave.
// =====================================================================
#define LDS_PAD 8
__global__ void __launch_bounds__(128)
wmma_gemm(const _Float16* __restrict__ Wh, const _Float16* __restrict__ XT,
          float* __restrict__ Y, int MP, int Kp, long Ncols, int cbCount)
{
  __shared__ __align__(16) _Float16 As[32 * (1152 + LDS_PAD)];
  int rb = blockIdx.x / cbCount;
  int cb = blockIdx.x % cbCount;
  int strideL = Kp + LDS_PAD;
  {
    int chunksPerRow = Kp >> 3;
    int tot = 32 * chunksPerRow;
    for (int i = threadIdx.x; i < tot; i += 128) {
      int r = i / chunksPerRow, cc = i % chunksPerRow;
      *(v8h*)(As + (size_t)r * strideL + (cc << 3)) =
          *(const v8h*)(Wh + ((size_t)(rb * 32 + r)) * Kp + (cc << 3));
    }
  }
  __syncthreads();
  int wv = threadIdx.x >> 5;
  int lane = threadIdx.x & 31;
  int l15 = lane & 15;
  int hi = (lane >> 4) << 3;
  long col0 = (long)cb * 256 + wv * 64;
  v8f acc[2][4] = {};
  const _Float16* a0 = As + (size_t)l15 * strideL + hi;
  const _Float16* a1 = As + (size_t)(16 + l15) * strideL + hi;
  for (int k0 = 0; k0 < Kp; k0 += 32) {
    Frag16 fa0, fa1;
    fa0.h[0] = *(const v8h*)(a0 + k0);
    fa0.h[1] = *(const v8h*)(a0 + k0 + 16);
    fa1.h[0] = *(const v8h*)(a1 + k0);
    fa1.h[1] = *(const v8h*)(a1 + k0 + 16);
#pragma unroll
    for (int t = 0; t < 4; ++t) {
      const _Float16* bRow = XT + (size_t)(col0 + t * 16 + l15) * Kp + hi;
      Frag16 fb;
      fb.h[0] = *(const v8h*)(bRow + k0);
      fb.h[1] = *(const v8h*)(bRow + k0 + 16);
      acc[0][t] = __builtin_amdgcn_wmma_f32_16x16x32_f16(
          false, fa0.v, false, fb.v, (short)0, acc[0][t], false, false);
      acc[1][t] = __builtin_amdgcn_wmma_f32_16x16x32_f16(
          false, fa1.v, false, fb.v, (short)0, acc[1][t], false, false);
    }
  }
  int rlane = (lane >> 4) << 3;
#pragma unroll
  for (int t2 = 0; t2 < 2; ++t2) {
    int rbase = rb * 32 + t2 * 16 + rlane;
#pragma unroll
    for (int t = 0; t < 4; ++t) {
      long col = col0 + t * 16 + l15;
#pragma unroll
      for (int r = 0; r < 8; ++r) {
        int row = rbase + r;
        if (row < MP) Y[(size_t)row * Ncols + col] = acc[t2][t][r];
      }
    }
  }
}

// ------------------- tiny GEMM with K=3 (raw / normal / curvature embed)
// layout 0: in[m*3+j]   layout 1: in[b*3*n + j*n + p], m=b*n+p
__global__ void k_lin3(const float* __restrict__ W, const float* __restrict__ in,
                       float* __restrict__ Y, int C, long M, int n, int layout)
{
  long idx = (long)blockIdx.x * blockDim.x + threadIdx.x;
  long total = (long)C * M;
  if (idx >= total) return;
  int c = (int)(idx / M);
  long m = idx % M;
  float a0, a1, a2;
  if (layout == 0) {
    a0 = in[m * 3]; a1 = in[m * 3 + 1]; a2 = in[m * 3 + 2];
  } else {
    long b = m / n, p = m % n;
    const float* base = in + (size_t)b * 3 * n + p;
    a0 = base[0]; a1 = base[n]; a2 = base[2 * n];
  }
  Y[idx] = W[c * 3] * a0 + W[c * 3 + 1] * a1 + W[c * 3 + 2] * a2;
}

// ------------------- per-channel BN statistics (block per channel)
__global__ void k_bn_stats(const float* __restrict__ Y, float* __restrict__ mu,
                           float* __restrict__ var, long M)
{
  int c = blockIdx.x;
  const float* row = Y + (size_t)c * M;
  float s = 0.f, sq = 0.f;
  for (long i = threadIdx.x; i < M; i += blockDim.x) {
    float v = row[i]; s += v; sq += v * v;
  }
  __shared__ float sS[256], sQ[256];
  sS[threadIdx.x] = s; sQ[threadIdx.x] = sq; __syncthreads();
  for (int o = blockDim.x >> 1; o > 0; o >>= 1) {
    if ((int)threadIdx.x < o) { sS[threadIdx.x] += sS[threadIdx.x + o]; sQ[threadIdx.x] += sQ[threadIdx.x + o]; }
    __syncthreads();
  }
  if (threadIdx.x == 0) {
    float m = sS[0] / (float)M;
    mu[c] = m; var[c] = sQ[0] / (float)M - m * m;
  }
}

__global__ void k_bn_apply(const float* __restrict__ Y, float* __restrict__ dst,
                           const float* __restrict__ g, const float* __restrict__ bt,
                           const float* __restrict__ mu, const float* __restrict__ var,
                           int C, long M, int relu)
{
  long idx = (long)blockIdx.x * blockDim.x + threadIdx.x;
  long total = (long)C * M;
  if (idx >= total) return;
  int c = (int)(idx / M);
  float v = g[c] * (Y[idx] - mu[c]) * rsqrtf(var[c] + EPS_BN) + bt[c];
  if (relu) v = fmaxf(v, 0.f);
  dst[idx] = v;
}

// ------------------- farthest point sampling (block per batch, dist in LDS)
__global__ void k_fps(const float* __restrict__ xyz, int n, int npoint, int* __restrict__ fidx)
{
  int b = blockIdx.x;
  const float* P = xyz + (size_t)b * n * 3;
  __shared__ float dist[2048];
  __shared__ float rv[256];
  __shared__ int ri[256];
  __shared__ int sFar;
  for (int i = threadIdx.x; i < n; i += blockDim.x) dist[i] = 1e10f;
  if (threadIdx.x == 0) sFar = 0;
  __syncthreads();
  for (int s = 0; s < npoint; ++s) {
    int far = sFar;
    if (threadIdx.x == 0) fidx[b * npoint + s] = far;
    float cx = P[far * 3], cy = P[far * 3 + 1], cz = P[far * 3 + 2];
    float bV = -1.f; int bI = 0x7fffffff;
    for (int i = threadIdx.x; i < n; i += blockDim.x) {
      float dx = P[i * 3] - cx, dy = P[i * 3 + 1] - cy, dz = P[i * 3 + 2] - cz;
      float d = dx * dx + dy * dy + dz * dz;
      float nd = fminf(dist[i], d);
      dist[i] = nd;
      if (nd > bV || (nd == bV && i < bI)) { bV = nd; bI = i; }
    }
    rv[threadIdx.x] = bV; ri[threadIdx.x] = bI; __syncthreads();
    for (int o = blockDim.x >> 1; o > 0; o >>= 1) {
      if ((int)threadIdx.x < o) {
        if (rv[threadIdx.x + o] > rv[threadIdx.x] ||
            (rv[threadIdx.x + o] == rv[threadIdx.x] && ri[threadIdx.x + o] < ri[threadIdx.x])) {
          rv[threadIdx.x] = rv[threadIdx.x + o]; ri[threadIdx.x] = ri[threadIdx.x + o];
        }
      }
      __syncthreads();
    }
    if (threadIdx.x == 0) sFar = ri[0];
    __syncthreads();
  }
}

__global__ void k_gather_xyz(const float* __restrict__ xyz, const int* __restrict__ fidx,
                             float* __restrict__ dst, int gn, int n)
{
  long idx = (long)blockIdx.x * blockDim.x + threadIdx.x;
  long total = (long)NB * gn * 3;
  if (idx >= total) return;
  int d = (int)(idx % 3);
  long bg = idx / 3;
  int b = (int)(bg / gn);
  dst[idx] = xyz[((size_t)b * n + fidx[bg]) * 3 + d];
}

// ------------------- exact KNN, 40 arg-min passes (block per query)
__global__ void k_knn(const float* __restrict__ xyz, const float* __restrict__ q,
                      int* __restrict__ kidx, int n, int gn)
{
  long bq = blockIdx.x;
  int b = (int)(bq / gn);
  const float* P = xyz + (size_t)b * n * 3;
  float qx = q[bq * 3], qy = q[bq * 3 + 1], qz = q[bq * 3 + 2];
  float q2 = qx * qx + qy * qy + qz * qz;
  __shared__ float d2[2048];
  __shared__ float rv[128];
  __shared__ int ri[128];
  for (int i = threadIdx.x; i < n; i += blockDim.x) {
    float px = P[i * 3], py = P[i * 3 + 1], pz = P[i * 3 + 2];
    d2[i] = q2 + px * px + py * py + pz * pz - 2.f * (qx * px + qy * py + qz * pz);
  }
  __syncthreads();
  for (int s = 0; s < KK; ++s) {
    float bV = 3.4e38f; int bI = 0x7fffffff;
    for (int i = threadIdx.x; i < n; i += blockDim.x) {
      float v = d2[i];
      if (v < bV || (v == bV && i < bI)) { bV = v; bI = i; }
    }
    rv[threadIdx.x] = bV; ri[threadIdx.x] = bI; __syncthreads();
    for (int o = blockDim.x >> 1; o > 0; o >>= 1) {
      if ((int)threadIdx.x < o) {
        if (rv[threadIdx.x + o] < rv[threadIdx.x] ||
            (rv[threadIdx.x + o] == rv[threadIdx.x] && ri[threadIdx.x + o] < ri[threadIdx.x])) {
          rv[threadIdx.x] = rv[threadIdx.x + o]; ri[threadIdx.x] = ri[threadIdx.x + o];
        }
      }
      __syncthreads();
    }
    if (threadIdx.x == 0) { kidx[bq * KK + s] = ri[0]; d2[ri[0]] = 3.4e38f; }
    __syncthreads();
  }
}

// ------------------- 3x3 covariance eigen (normal + curvature)
__global__ void k_geo(const float* __restrict__ xyz, const int* __restrict__ kidx,
                      float* __restrict__ nrm, float* __restrict__ cv, int gn, int n, long total)
{
  long bq = (long)blockIdx.x * blockDim.x + threadIdx.x;
  if (bq >= total) return;
  int b = (int)(bq / gn);
  const float* P = xyz + (size_t)b * n * 3;
  const int* ki = kidx + bq * KK;
  float mx = 0, my = 0, mz = 0, xx = 0, xy = 0, xz = 0, yy = 0, yz = 0, zz = 0;
  for (int k = 0; k < KS; ++k) {
    int id = ki[k];
    float x = P[id * 3], y = P[id * 3 + 1], z = P[id * 3 + 2];
    mx += x; my += y; mz += z;
    xx += x * x; xy += x * y; xz += x * z; yy += y * y; yz += y * z; zz += z * z;
  }
  float inv = 1.f / (float)KS;
  mx *= inv; my *= inv; mz *= inv;
  float c00 = xx * inv - mx * mx, c01 = xy * inv - mx * my, c02 = xz * inv - mx * mz;
  float c11 = yy * inv - my * my, c12 = yz * inv - my * mz, c22 = zz * inv - mz * mz;
  float p1 = c01 * c01 + c02 * c02 + c12 * c12;
  float qm = (c00 + c11 + c22) / 3.f;
  float e1, e2, e3;
  if (p1 < 1e-20f) {
    e1 = fmaxf(c00, fmaxf(c11, c22));
    e3 = fminf(c00, fminf(c11, c22));
    e2 = c00 + c11 + c22 - e1 - e3;
  } else {
    float p2 = (c00 - qm) * (c00 - qm) + (c11 - qm) * (c11 - qm) + (c22 - qm) * (c22 - qm) + 2.f * p1;
    float pp = sqrtf(p2 / 6.f);
    float ip = 1.f / pp;
    float b00 = (c00 - qm) * ip, b01 = c01 * ip, b02 = c02 * ip;
    float b11 = (c11 - qm) * ip, b12 = c12 * ip, b22 = (c22 - qm) * ip;
    float detB = b00 * (b11 * b22 - b12 * b12) - b01 * (b01 * b22 - b12 * b02) + b02 * (b01 * b12 - b11 * b02);
    float r = fminf(1.f, fmaxf(-1.f, detB * 0.5f));
    float phi = acosf(r) / 3.f;
    e1 = qm + 2.f * pp * cosf(phi);
    e3 = qm + 2.f * pp * cosf(phi + 2.0943951023931953f);
    e2 = 3.f * qm - e1 - e3;
  }
  float a00 = c00 - e3, a11 = c11 - e3, a22 = c22 - e3;
  float v0x = c01 * c12 - a11 * c02, v0y = c02 * c01 - a00 * c12, v0z = a00 * a11 - c01 * c01;
  float v1x = c01 * a22 - c12 * c02, v1y = c02 * c02 - a00 * a22, v1z = a00 * c12 - c01 * c02;
  float v2x = a11 * a22 - c12 * c12, v2y = c12 * c02 - c01 * a22, v2z = c01 * c12 - a11 * c02;
  float n0 = v0x * v0x + v0y * v0y + v0z * v0z;
  float n1 = v1x * v1x + v1y * v1y + v1z * v1z;
  float n2 = v2x * v2x + v2y * v2y + v2z * v2z;
  float vx = v0x, vy = v0y, vz = v0z, nn = n0;
  if (n1 > nn) { vx = v1x; vy = v1y; vz = v1z; nn = n1; }
  if (n2 > nn) { vx = v2x; vy = v2y; vz = v2z; nn = n2; }
  if (nn < 1e-30f) { vx = 0.f; vy = 0.f; vz = 1.f; nn = 1.f; }
  float rn = rsqrtf(nn);
  nrm[bq * 3] = vx * rn; nrm[bq * 3 + 1] = vy * rn; nrm[bq * 3 + 2] = vz * rn;
  float s1 = e1 + 1e-9f, s2 = e2 + 1e-9f, s3 = e3 + 1e-9f;
  float si = 1.f / (s1 + s2 + s3);
  cv[bq * 3] = s1 * si; cv[bq * 3 + 1] = s2 * si; cv[bq * 3 + 2] = s3 * si;
}

// ------------------- global sum/sumsq of d = knn_xyz - lc_xyz
__global__ void k_dstats(const float* __restrict__ xyz, const float* __restrict__ lc,
                         const int* __restrict__ kidx, float* __restrict__ acc,
                         int gn, int n, long total)
{
  float s = 0.f, sq = 0.f;
  for (long i = (long)blockIdx.x * blockDim.x + threadIdx.x; i < total;
       i += (long)gridDim.x * blockDim.x) {
    int d = (int)(i % 3);
    long t = i / 3;
    int k = (int)(t % KK);
    long bg = t / KK;
    int b = (int)(bg / gn);
    int id = kidx[bg * KK + k];
    float v = xyz[((size_t)b * n + id) * 3 + d] - lc[bg * 3 + d];
    s += v; sq += v * v;
  }
  __shared__ float sS[256], sQ[256];
  sS[threadIdx.x] = s; sQ[threadIdx.x] = sq; __syncthreads();
  for (int o = blockDim.x >> 1; o > 0; o >>= 1) {
    if ((int)threadIdx.x < o) { sS[threadIdx.x] += sS[threadIdx.x + o]; sQ[threadIdx.x] += sQ[threadIdx.x + o]; }
    __syncthreads();
  }
  if (threadIdx.x == 0) { atomicAdd(&acc[0], sS[0]); atomicAdd(&acc[1], sQ[0]); }
}

// ------------------- build KFT (M x Kp, f16): [knn_x | lc_x | 0-pad]
__global__ void k_build_kft(const float* __restrict__ x, const int* __restrict__ kidx,
                            const int* __restrict__ fidx, _Float16* __restrict__ KFT,
                            int gn, int n, int C, int Kp, long M)
{
  long idx = (long)blockIdx.x * blockDim.x + threadIdx.x;
  long total = M * Kp;
  if (idx >= total) return;
  int col = (int)(idx % Kp);
  long m = idx / Kp;
  int k = (int)(m % KK);
  long bg = m / KK;
  int b = (int)(bg / gn);
  float v = 0.f;
  if (col < C) {
    int id = kidx[bg * KK + k];
    v = x[(size_t)col * (NB * n) + (size_t)b * n + id];
  } else if (col < 2 * C) {
    int id = fidx[bg];
    v = x[(size_t)(col - C) * (NB * n) + (size_t)b * n + id];
  }
  KFT[idx] = (_Float16)v;
}

// ------------------- f32 weight -> padded f16 (rowsP x Kp)
__global__ void k_convert_w(const float* __restrict__ W, _Float16* __restrict__ Wh,
                            int rows, int cols, int rowsP, int Kp)
{
  long idx = (long)blockIdx.x * blockDim.x + threadIdx.x;
  long total = (long)rowsP * Kp;
  if (idx >= total) return;
  int r = (int)(idx / Kp), c = (int)(idx % Kp);
  Wh[idx] = (r < rows && c < cols) ? (_Float16)W[(size_t)r * cols + c] : (_Float16)0.f;
}

// ------------------- fused: BN+relu(lin1) -> positional encode -> f16 wT
__global__ void k_pose(const float* __restrict__ Y, const float* __restrict__ g,
                       const float* __restrict__ bt, const float* __restrict__ mu,
                       const float* __restrict__ var, const float* __restrict__ xyz,
                       const float* __restrict__ lc, const int* __restrict__ kidx,
                       const float* __restrict__ dacc, _Float16* __restrict__ WT,
                       int outC, long M, int Kp, int gn, int n, int fd)
{
  long idx = (long)blockIdx.x * blockDim.x + threadIdx.x;
  long total = M * Kp;
  if (idx >= total) return;
  int col = (int)(idx % Kp);
  long m = idx / Kp;
  if (col >= outC) { WT[idx] = (_Float16)0.f; return; }
  int c = col;
  float y = Y[(size_t)c * M + m];
  float kf = fmaxf(g[c] * (y - mu[c]) * rsqrtf(var[c] + EPS_BN) + bt[c], 0.f);
  int k = (int)(m % KK);
  long bg = m / KK;
  int b = (int)(bg / gn);
  int d3 = c / (2 * fd);
  int j = c % (2 * fd);
  int jf = (j < fd) ? j : j - fd;
  int id = kidx[bg * KK + k];
  float dval = xyz[((size_t)b * n + id) * 3 + d3] - lc[bg * 3 + d3];
  float cnt = (float)(3 * M);
  float sm = dacc[0], sqm = dacc[1];
  float mean = sm / cnt;
  float vD = (sqm - sm * mean) / (cnt - 1.f);
  float istd = 1.f / (sqrtf(fmaxf(vD, 0.f)) + 1e-5f);
  float freq = powf(1000.f, (float)jf / (float)fd);
  float dv = 100.f * dval * istd / freq;
  float pe = (j < fd) ? sinf(dv) : cosf(dv);
  WT[idx] = (_Float16)((kf + pe) * pe);
}

// ------------------- BN+relu of W1 output -> padded f16 transposed h
__global__ void k_happly(const float* __restrict__ Y, const float* __restrict__ g,
                         const float* __restrict__ bt, const float* __restrict__ mu,
                         const float* __restrict__ var, _Float16* __restrict__ HT,
                         int halfC, long M, int KpH)
{
  long idx = (long)blockIdx.x * blockDim.x + threadIdx.x;
  long total = M * KpH;
  if (idx >= total) return;
  int col = (int)(idx % KpH);
  long m = idx / KpH;
  if (col >= halfC) { HT[idx] = (_Float16)0.f; return; }
  float y = Y[(size_t)col * M + m];
  HT[idx] = (_Float16)fmaxf(g[col] * (y - mu[col]) * rsqrtf(var[col] + EPS_BN) + bt[col], 0.f);
}

// ------------------- fused: residual relu(bn(h2)+w), max+mean over K
__global__ void k_lc(const float* __restrict__ Y2, const _Float16* __restrict__ WT,
                     const float* __restrict__ g, const float* __restrict__ bt,
                     const float* __restrict__ mu, const float* __restrict__ var,
                     float* __restrict__ lcb, int outC, long BG, int Kp)
{
  long idx = (long)blockIdx.x * blockDim.x + threadIdx.x;
  long total = (long)outC * BG;
  if (idx >= total) return;
  int c = (int)(idx / BG);
  long bg = idx % BG;
  float iv = rsqrtf(var[c] + EPS_BN), gg = g[c], bb = bt[c], mm = mu[c];
  const float* yrow = Y2 + (size_t)c * (BG * KK) + bg * KK;
  const _Float16* wcol = WT + (size_t)bg * KK * Kp + c;
  float mx = -3.4e38f, sum = 0.f;
  for (int k = 0; k < KK; ++k) {
    float v = gg * (yrow[k] - mm) * iv + bb + (float)wcol[(size_t)k * Kp];
    v = fmaxf(v, 0.f);
    mx = fmaxf(mx, v);
    sum += v;
  }
  lcb[idx] = mx + sum / (float)KK;
}

// ------------------- sum_s alphas*feats+betas  -> f16 transposed (BG x Kp)
__global__ void k_combine(const float* __restrict__ lcb, const float* __restrict__ ne,
                          const float* __restrict__ ce, const float* __restrict__ alphas,
                          const float* __restrict__ betas, _Float16* __restrict__ ST,
                          int outC, long BG, int Kp)
{
  long idx = (long)blockIdx.x * blockDim.x + threadIdx.x;
  long total = BG * Kp;
  if (idx >= total) return;
  int col = (int)(idx % Kp);
  long bg = idx / Kp;
  if (col >= outC) { ST[idx] = (_Float16)0.f; return; }
  int c = col;
  float v = alphas[c] * lcb[(size_t)c * BG + bg] + betas[c]
          + alphas[outC + c] * ne[(size_t)c * BG + bg] + betas[outC + c]
          + alphas[2 * outC + c] * ce[(size_t)c * BG + bg] + betas[2 * outC + c];
  ST[idx] = (_Float16)v;
}

// ------------------- (C, B*n) -> (B, C, n) into d_out
__global__ void k_store_x(const float* __restrict__ X, float* __restrict__ dst, int C, int n)
{
  long idx = (long)blockIdx.x * blockDim.x + threadIdx.x;
  long total = (long)NB * C * n;
  if (idx >= total) return;
  int p = (int)(idx % n);
  long t = idx / n;
  int c = (int)(t % C);
  int b = (int)(t / C);
  dst[idx] = X[(size_t)c * NB * n + (size_t)b * n + p];
}

static inline long cdivl(long a, long b) { return (a + b - 1) / b; }
static inline int rup32(int v) { return ((v + 31) / 32) * 32; }

extern "C" void kernel_launch(void* const* d_in, const int* in_sizes, int n_in,
                              void* d_out, int out_size, void* d_ws, size_t ws_size,
                              hipStream_t stream)
{
  (void)in_sizes; (void)n_in; (void)out_size;
  char* wsB = (char*)d_ws;
  size_t off = 0;
  auto take = [&](size_t bytes) -> char* {
    char* p = wsB + off;
    off += (bytes + 255) & ~(size_t)255;
    return p;
  };
  _Float16* A16 = (_Float16*)take(104857600);   // KFT / hT (f16, M x Kpad)
  _Float16* B16 = (_Float16*)take(104857600);   // wT (f16, M x Kpad)
  float*    Cf  = (float*)take(188743680);      // GEMM f32 accumulator (out x M)
  float*    xA  = (float*)take(4718592);
  float*    xB  = (float*)take(4718592);
  int*      fidx = (int*)take(32768);
  int*      kidx = (int*)take(1310720);
  float*    nbuf = (float*)take(98304);
  float*    cvbuf = (float*)take(98304);
  float*    nemb = (float*)take(4718592);
  float*    cemb = (float*)take(4718592);
  float*    lcb  = (float*)take(4718592);
  _Float16* ST16 = (_Float16*)take(2621440);
  _Float16* Wh16 = (_Float16*)take(2654208);
  float*    mu   = (float*)take(4608);
  float*    var  = (float*)take(4608);
  float*    dacc = (float*)take(256);
  if (off > ws_size) return;

  float* outF = (float*)d_out;
  const long xyzOff[5] = {0, 49152, 73728, 86016, 92160};
  const long xOff[5]   = {95232, 1274880, 2454528, 3634176, 4813824};

  auto gemm = [&](const _Float16* Wf, const _Float16* XT, float* Y,
                  int MPreal, int Kp, long N) {
    int cbCount = (int)(N / 256);
    int blocks = (rup32(MPreal) / 32) * cbCount;
    wmma_gemm<<<blocks, 128, 0, stream>>>(Wf, XT, Y, MPreal, Kp, N, cbCount);
  };

  // xyz_list[0] = input xyz
  hipMemcpyAsync(outF, d_in[0], 49152 * sizeof(float), hipMemcpyDeviceToDevice, stream);

  // ---- raw lin1: (B,3,2048) -> (72, B*2048), BN over (B, N), relu
  {
    const float* xin = (const float*)d_in[1];
    const float* rW = (const float*)d_in[2];
    const float* rB = (const float*)d_in[3];
    const float* rG = (const float*)d_in[4];
    long M0 = (long)NB * 2048;
    k_lin3<<<(unsigned)cdivl(72 * M0, 256), 256, 0, stream>>>(rW, xin, Cf, 72, M0, 2048, 1);
    k_bn_stats<<<72, 256, 0, stream>>>(Cf, mu, var, M0);
    k_bn_apply<<<(unsigned)cdivl(72 * M0, 256), 256, 0, stream>>>(Cf, xA, rG, rB, mu, var, 72, M0, 1);
    k_store_x<<<(unsigned)cdivl(72 * M0, 256), 256, 0, stream>>>(xA, outF + xOff[0], 72, 2048);
  }

  int n = 2048, Cprev = 72;
  float* xcur = xA;
  float* xnxt = xB;

  for (int i = 0; i < 4; ++i) {
    int out = Cprev * 2;
    int gn = n / 2;
    long BG = (long)NB * gn;
    long M = BG * KK;
    int Kp = rup32(out);
    int half = out / 2;
    int KpH = rup32(half);
    int fd = out / 6;
    int base = 5 + 22 * i;
    const float* W1 = (const float*)d_in[base + 0];
    const float* W2 = (const float*)d_in[base + 1];
    const float* bt1 = (const float*)d_in[base + 4];
    const float* bt2 = (const float*)d_in[base + 5];
    const float* g1 = (const float*)d_in[base + 6];
    const float* g2 = (const float*)d_in[base + 7];
    const float* cW = (const float*)d_in[base + 8];
    const float* cB = (const float*)d_in[base + 9];
    const float* cG = (const float*)d_in[base + 10];
    const float* l1W = (const float*)d_in[base + 11];
    const float* l1B = (const float*)d_in[base + 12];
    const float* l1G = (const float*)d_in[base + 13];
    const float* alphas = (const float*)d_in[base + 14];
    const float* betas = (const float*)d_in[base + 15];
    const float* mW = (const float*)d_in[base + 16];
    const float* mB = (const float*)d_in[base + 17];
    const float* mG = (const float*)d_in[base + 18];
    const float* nW = (const float*)d_in[base + 19];
    const float* nBb = (const float*)d_in[base + 20];
    const float* nG = (const float*)d_in[base + 21];

    float* xyzC = outF + xyzOff[i];
    float* xyzN = outF + xyzOff[i + 1];

    // sampling / grouping / geometry
    k_fps<<<NB, 256, 0, stream>>>(xyzC, n, gn, fidx);
    k_gather_xyz<<<(unsigned)cdivl(BG * 3, 256), 256, 0, stream>>>(xyzC, fidx, xyzN, gn, n);
    k_knn<<<(unsigned)BG, 128, 0, stream>>>(xyzC, xyzN, kidx, n, gn);
    k_geo<<<(unsigned)cdivl(BG, 128), 128, 0, stream>>>(xyzC, kidx, nbuf, cvbuf, gn, n, BG);

    // normal / curvature embeddings (K=3, plain VALU path)
    k_lin3<<<(unsigned)cdivl((long)out * BG, 256), 256, 0, stream>>>(nW, nbuf, nemb, out, BG, 0, 0);
    k_bn_stats<<<out, 256, 0, stream>>>(nemb, mu, var, BG);
    k_bn_apply<<<(unsigned)cdivl((long)out * BG, 256), 256, 0, stream>>>(nemb, nemb, nG, nBb, mu, var, out, BG, 1);
    k_lin3<<<(unsigned)cdivl((long)out * BG, 256), 256, 0, stream>>>(cW, cvbuf, cemb, out, BG, 0, 0);
    k_bn_stats<<<out, 256, 0, stream>>>(cemb, mu, var, BG);
    k_bn_apply<<<(unsigned)cdivl((long)out * BG, 256), 256, 0, stream>>>(cemb, cemb, cG, cB, mu, var, out, BG, 1);

    // global std of d (ddof=1)
    hipMemsetAsync(dacc, 0, 8, stream);
    k_dstats<<<1024, 256, 0, stream>>>(xyzC, xyzN, kidx, dacc, gn, n, 3 * M);

    // KFT and lin1 GEMM (WMMA)
    k_build_kft<<<(unsigned)cdivl(M * Kp, 256), 256, 0, stream>>>(xcur, kidx, fidx, A16, gn, n, Cprev, Kp, M);
    k_convert_w<<<(unsigned)cdivl((long)rup32(out) * Kp, 256), 256, 0, stream>>>(l1W, Wh16, out, out, rup32(out), Kp);
    gemm(Wh16, A16, Cf, out, Kp, M);
    k_bn_stats<<<out, 256, 0, stream>>>(Cf, mu, var, M);
    k_pose<<<(unsigned)cdivl(M * Kp, 256), 256, 0, stream>>>(Cf, l1G, l1B, mu, var, xyzC, xyzN, kidx, dacc,
                                                             B16, out, M, Kp, gn, n, fd);

    // block: W1 GEMM -> BN+relu -> hT
    k_convert_w<<<(unsigned)cdivl((long)rup32(half) * Kp, 256), 256, 0, stream>>>(W1, Wh16, half, out, rup32(half), Kp);
    gemm(Wh16, B16, Cf, half, Kp, M);
    k_bn_stats<<<half, 256, 0, stream>>>(Cf, mu, var, M);
    k_happly<<<(unsigned)cdivl(M * KpH, 256), 256, 0, stream>>>(Cf, g1, bt1, mu, var, A16, half, M, KpH);

    // block: W2 GEMM -> BN -> residual+relu -> max+mean over K
    k_convert_w<<<(unsigned)cdivl((long)rup32(out) * KpH, 256), 256, 0, stream>>>(W2, Wh16, out, half, rup32(out), KpH);
    gemm(Wh16, A16, Cf, out, KpH, M);
    k_bn_stats<<<out, 256, 0, stream>>>(Cf, mu, var, M);
    k_lc<<<(unsigned)cdivl((long)out * BG, 256), 256, 0, stream>>>(Cf, B16, g2, bt2, mu, var, lcb, out, BG, Kp);

    // MAA aggregation + final lin (WMMA) + BN+relu
    k_combine<<<(unsigned)cdivl(BG * Kp, 256), 256, 0, stream>>>(lcb, nemb, cemb, alphas, betas, ST16, out, BG, Kp);
    k_convert_w<<<(unsigned)cdivl((long)rup32(out) * Kp, 256), 256, 0, stream>>>(mW, Wh16, out, out, rup32(out), Kp);
    gemm(Wh16, ST16, Cf, out, Kp, BG);
    k_bn_stats<<<out, 256, 0, stream>>>(Cf, mu, var, BG);
    k_bn_apply<<<(unsigned)cdivl((long)out * BG, 256), 256, 0, stream>>>(Cf, xnxt, mG, mB, mu, var, out, BG, 1);
    k_store_x<<<(unsigned)cdivl((long)out * BG, 256), 256, 0, stream>>>(xnxt, outF + xOff[i + 1], out, gn);

    float* t = xcur; xcur = xnxt; xnxt = t;
    n = gn;
    Cprev = out;
  }
}
